// TriTransLayer_55628416418474
// MI455X (gfx1250) — matrix operand order repaired and checked by reading, not compiled, not run on hardware
//
#include <hip/hip_runtime.h>

typedef __attribute__((ext_vector_type(16))) _Float16 v16h;
typedef __attribute__((ext_vector_type(8)))  float    v8f;

// Problem constants (from reference: B=2, N=64, E=128, H=8, D=16)
#define NB   2
#define NN   64

// ---- workspace layout (bytes) ----
#define OFF_WQ   ((size_t)0)
#define OFF_WKE  ((size_t)32768)
#define OFF_WV   ((size_t)65536)
#define OFF_WG   ((size_t)98304)
#define OFF_WKA  ((size_t)131072)
#define OFF_WVA  ((size_t)147456)
#define OFF_Q    ((size_t)163840)
#define OFF_KE   (OFF_Q    + (size_t)4194304)
#define OFF_VE   (OFF_KE   + (size_t)4194304)
#define OFF_GATE (OFF_VE   + (size_t)4194304)

__device__ __forceinline__ float fast_sigmoid(float x) {
    return __builtin_amdgcn_rcpf(1.0f + __expf(-x));
}

// ------------------------------------------------------------------
// Kernel 1: convert f32 weight (K x 128) into f16, pre-swizzled into
// WMMA B-fragment order. Fragment f = et*ktiles + kt, each lane's 16
// halves contiguous:  dst[((f*32 + lane)*16) + t]
//   lane<16 : B[kt*32 +   t + (t&8)][et*16 + lane]
//   lane>=16: B[kt*32 + 8+t + (t&8)][et*16 + lane-16]
// ------------------------------------------------------------------
__global__ void swizzle_w_kernel(const float* __restrict__ src,
                                 _Float16* __restrict__ dst,
                                 int ktiles, int n) {
    int u = blockIdx.x * blockDim.x + threadIdx.x;
    if (u >= n) return;
    int t    = u & 15;
    int lane = (u >> 4) & 31;
    int f    = u >> 9;
    int kt   = f % ktiles;
    int et   = f / ktiles;
    int c    = kt * 32 + ((lane >> 4) & 1) * 8 + t + (t & 8);
    int e    = et * 16 + (lane & 15);
    dst[u] = (_Float16)src[c * 128 + e];
}

// A-fragment loader from LDS (row-major f16 with stride strideH halves)
__device__ __forceinline__ v16h load_a_frag(const _Float16* sh, int strideH,
                                            int m, int kbase0) {
    const _Float16* p = sh + m * strideH + kbase0;
    v16h a;
#pragma unroll
    for (int i = 0; i < 8; ++i) { a[i] = p[i]; a[i + 8] = p[i + 16]; }
    return a;
}

// ------------------------------------------------------------------
// Kernel 2: LayerNorm + 4 projections (q*0.25, ke, ve, sigmoid gate)
// 16 rows per workgroup; wave w handles projection w via WMMA.
// ------------------------------------------------------------------
__global__ __launch_bounds__(128)
void proj_kernel(const float* __restrict__ pair_emb,
                 const float* __restrict__ ln_g, const float* __restrict__ ln_b,
                 const _Float16* __restrict__ WqS,  const float* __restrict__ bq,
                 const _Float16* __restrict__ WkeS, const float* __restrict__ bke,
                 const _Float16* __restrict__ WvS,  const float* __restrict__ bv,
                 const _Float16* __restrict__ WgS,  const float* __restrict__ bg,
                 float* __restrict__ qWS, float* __restrict__ keWS,
                 float* __restrict__ veWS, float* __restrict__ gateWS) {
    __shared__ __align__(16) _Float16 pe16[16 * 136];
    __shared__ float red[16][8], red2[16][8];
    __shared__ float meansh[16], rstdsh[16];

    const int t    = threadIdx.x;
    const int lane = t & 31;
    // wave index is uniform per wave -> force into SGPR
    const int w    = __builtin_amdgcn_readfirstlane(t >> 5);
    const int r    = t >> 3;     // row 0..15
    const int g    = t & 7;      // 8 groups of 16 elems
    const size_t rowbase = (size_t)blockIdx.x * 16;

    // ---- LayerNorm stats ----
    const float* xr = pair_emb + (rowbase + r) * 128 + g * 16;
    float xv[16];
    float s = 0.f, s2 = 0.f;
#pragma unroll
    for (int i = 0; i < 16; ++i) { float x = xr[i]; xv[i] = x; s += x; s2 += x * x; }
    red[r][g] = s; red2[r][g] = s2;
    __syncthreads();
    if (t < 16) {
        float ss = 0.f, ss2 = 0.f;
#pragma unroll
        for (int i = 0; i < 8; ++i) { ss += red[t][i]; ss2 += red2[t][i]; }
        float mean = ss * (1.0f / 128.0f);
        float var  = ss2 * (1.0f / 128.0f) - mean * mean;
        meansh[t] = mean;
        rstdsh[t] = rsqrtf(var + 1e-5f);
    }
    __syncthreads();
    {
        float mean = meansh[r], rstd = rstdsh[r];
#pragma unroll
        for (int i = 0; i < 16; ++i) {
            int e = g * 16 + i;
            float p = (xv[i] - mean) * rstd * ln_g[e] + ln_b[e];
            pe16[r * 136 + e] = (_Float16)p;
        }
    }
    __syncthreads();

    // ---- per-wave projection via WMMA (all selections scalar) ----
    const _Float16* Wsz; const float* bias; float* dst;
    switch (w) {
        case 0:  Wsz = WqS;  bias = bq;  dst = qWS;    break;
        case 1:  Wsz = WkeS; bias = bke; dst = keWS;   break;
        case 2:  Wsz = WvS;  bias = bv;  dst = veWS;   break;
        default: Wsz = WgS;  bias = bg;  dst = gateWS; break;
    }
    const float scale  = (w == 0) ? 0.25f : 1.0f;   // q * 1/sqrt(D)
    const int m15      = lane & 15;
    const int khalf    = (lane & 16) ? 8 : 0;
    const int rowoff   = (lane & 16) ? 8 : 0;

    for (int et = 0; et < 8; ++et) {
        v8f acc = {};
#pragma unroll
        for (int kt = 0; kt < 4; ++kt) {
            v16h a = load_a_frag(pe16, 136, m15, kt * 32 + khalf);
            v16h b = *(const v16h*)(Wsz + ((size_t)(et * 4 + kt) * 32 + lane) * 16);
            acc = __builtin_amdgcn_wmma_f32_16x16x32_f16(
                false, a, false, b, (short)0, acc, false, false);
        }
        int e = et * 16 + m15;
        float bv_ = bias[e];
        float vals[8];
#pragma unroll
        for (int rr = 0; rr < 8; ++rr) vals[rr] = (acc[rr] + bv_) * scale;
        if (w == 3) {  // scalar branch: whole wave takes it or not
#pragma unroll
            for (int rr = 0; rr < 8; ++rr) vals[rr] = fast_sigmoid(vals[rr]);
        }
#pragma unroll
        for (int rr = 0; rr < 8; ++rr)
            dst[(rowbase + rowoff + rr) * 128 + e] = vals[rr];
    }
}

// ------------------------------------------------------------------
// Kernel 3: one workgroup per (b,i,x). Fused ka/va WMMA + attention.
// ------------------------------------------------------------------
__global__ __launch_bounds__(128)
void attn_kernel(const float* __restrict__ tri_emb,
                 const float* __restrict__ pair_mask,
                 const _Float16* __restrict__ WkaS, const float* __restrict__ bka,
                 const _Float16* __restrict__ WvaS, const float* __restrict__ bva,
                 const float* __restrict__ qWS,  const float* __restrict__ keWS,
                 const float* __restrict__ veWS, const float* __restrict__ gateWS,
                 const float* __restrict__ Wo, const float* __restrict__ bo,
                 float* __restrict__ out) {
    __shared__ __align__(16) _Float16 Tsh[64 * 72];  // tri tile f16
    __shared__ float ksh[64 * 132];                  // k = ke + ka + bka
    __shared__ float vsh[64 * 132];                  // v = ve + va + bva
    __shared__ float qsh[128];
    __shared__ float ssh[8 * 64];                    // scores / attn
    __shared__ float oash[128];                      // attn @ v

    const int t    = threadIdx.x;
    const int lane = t & 31;
    const int w    = __builtin_amdgcn_readfirstlane(t >> 5);  // scalar wave id
    const int bix  = blockIdx.x;           // ((b*N)+i)*N + x
    const int xx   = bix & 63;
    const int bij  = bix >> 6;             // b*N + i
    const int bb   = bix >> 12;

    // stage tri_emb tile (64 j x 64 c) as f16
    const float* tri = tri_emb + (size_t)bix * 4096;
    for (int u = t; u < 4096; u += 128)
        Tsh[(u >> 6) * 72 + (u & 63)] = (_Float16)tri[u];
    qsh[t] = qWS[(size_t)bix * 128 + t];
    __syncthreads();

    const int m15    = lane & 15;
    const int khalf  = (lane & 16) ? 8 : 0;
    const int rowoff = (lane & 16) ? 8 : 0;

    // 64 output tiles: mat(2) x jt(4) x et(8); K=64 -> 2 WMMAs each.
    // tt is scalar (w is SGPR), so mat/jt/et + pointer selects stay scalar.
    for (int tt = w; tt < 64; tt += 4) {
        const int mat = tt >> 5;
        const int rr  = tt & 31;
        const int jt  = rr >> 3;
        const int et  = rr & 7;
        const _Float16* Wsz = mat ? WvaS : WkaS;
        v8f acc = {};
#pragma unroll
        for (int kt = 0; kt < 2; ++kt) {
            v16h a = load_a_frag(Tsh, 72, jt * 16 + m15, kt * 32 + khalf);
            v16h b = *(const v16h*)(Wsz + ((size_t)(et * 2 + kt) * 32 + lane) * 16);
            acc = __builtin_amdgcn_wmma_f32_16x16x32_f16(
                false, a, false, b, (short)0, acc, false, false);
        }
        const int e = et * 16 + m15;
        const float* erow = (mat ? veWS : keWS) + (size_t)bij * 64 * 128;
        const float biasv = (mat ? bva : bka)[e];
        float* dsh = mat ? vsh : ksh;
#pragma unroll
        for (int q8 = 0; q8 < 8; ++q8) {
            int j = jt * 16 + rowoff + q8;
            dsh[j * 132 + e] = acc[q8] + erow[j * 128 + e] + biasv;
        }
    }
    __syncthreads();

    // scores[h,j] = q[h,:] . k[j, h*16:...] + (1-mask[b,j,x])*1e-9
#pragma unroll
    for (int s = 0; s < 4; ++s) {
        int idx = t * 4 + s;
        int h = idx >> 6, j = idx & 63;
        float maskv = pair_mask[(bb * 64 + j) * 64 + xx];
        float sc = (1.0f - maskv) * 1e-9f;
        const float* qp = qsh + h * 16;
        const float* kp = ksh + j * 132 + h * 16;
#pragma unroll
        for (int d = 0; d < 16; ++d) sc = fmaf(qp[d], kp[d], sc);
        ssh[idx] = sc;
    }
    __syncthreads();

    // softmax over j per head
    if (t < 8) {
        float mx = -3.4e38f;
#pragma unroll 4
        for (int j = 0; j < 64; ++j) mx = fmaxf(mx, ssh[t * 64 + j]);
        float sum = 0.f;
#pragma unroll 4
        for (int j = 0; j < 64; ++j) {
            float e_ = __expf(ssh[t * 64 + j] - mx);
            ssh[t * 64 + j] = e_; sum += e_;
        }
        float inv = __builtin_amdgcn_rcpf(sum);
#pragma unroll 4
        for (int j = 0; j < 64; ++j) ssh[t * 64 + j] *= inv;
    }
    __syncthreads();

    // out_attn[e] = sum_j attn[h(e),j] * v[j,e]
    {
        int h = t >> 4;
        float acc = 0.f;
#pragma unroll 8
        for (int j = 0; j < 64; ++j)
            acc = fmaf(ssh[h * 64 + j], vsh[j * 132 + t], acc);
        oash[t] = acc;
    }
    __syncthreads();

    // final: (out_attn @ Wo + bo) * gate
    {
        float acc = bo[t];
#pragma unroll 8
        for (int c = 0; c < 128; ++c)
            acc = fmaf(oash[c], Wo[c * 128 + t], acc);
        float gate = gateWS[(size_t)bix * 128 + t];
        out[(size_t)bix * 128 + t] = acc * gate;
    }
}

extern "C" void kernel_launch(void* const* d_in, const int* in_sizes, int n_in,
                              void* d_out, int out_size, void* d_ws, size_t ws_size,
                              hipStream_t stream) {
    const float* pair_emb  = (const float*)d_in[0];
    const float* tri_emb   = (const float*)d_in[1];
    const float* pair_mask = (const float*)d_in[2];
    const float* ln_g = (const float*)d_in[3];
    const float* ln_b = (const float*)d_in[4];
    const float* Wq  = (const float*)d_in[5];  const float* bq  = (const float*)d_in[6];
    const float* Wke = (const float*)d_in[7];  const float* bke = (const float*)d_in[8];
    const float* Wka = (const float*)d_in[9];  const float* bka = (const float*)d_in[10];
    const float* Wv  = (const float*)d_in[11]; const float* bv  = (const float*)d_in[12];
    const float* Wva = (const float*)d_in[13]; const float* bva = (const float*)d_in[14];
    const float* Wo  = (const float*)d_in[15]; const float* bo  = (const float*)d_in[16];
    const float* Wg  = (const float*)d_in[17]; const float* bg  = (const float*)d_in[18];

    char* ws = (char*)d_ws;
    _Float16* WqS  = (_Float16*)(ws + OFF_WQ);
    _Float16* WkeS = (_Float16*)(ws + OFF_WKE);
    _Float16* WvS  = (_Float16*)(ws + OFF_WV);
    _Float16* WgS  = (_Float16*)(ws + OFF_WG);
    _Float16* WkaS = (_Float16*)(ws + OFF_WKA);
    _Float16* WvaS = (_Float16*)(ws + OFF_WVA);
    float* qWS    = (float*)(ws + OFF_Q);
    float* keWS   = (float*)(ws + OFF_KE);
    float* veWS   = (float*)(ws + OFF_VE);
    float* gateWS = (float*)(ws + OFF_GATE);

    // 1) weight prep: f16 + B-fragment swizzle
    swizzle_w_kernel<<<64, 256, 0, stream>>>(Wq,  WqS,  4, 16384);
    swizzle_w_kernel<<<64, 256, 0, stream>>>(Wke, WkeS, 4, 16384);
    swizzle_w_kernel<<<64, 256, 0, stream>>>(Wv,  WvS,  4, 16384);
    swizzle_w_kernel<<<64, 256, 0, stream>>>(Wg,  WgS,  4, 16384);
    swizzle_w_kernel<<<32, 256, 0, stream>>>(Wka, WkaS, 2, 8192);
    swizzle_w_kernel<<<32, 256, 0, stream>>>(Wva, WvaS, 2, 8192);

    // 2) LN + projections (q, ke, ve, gate), 8192 rows / 16 per WG
    proj_kernel<<<512, 128, 0, stream>>>(pair_emb, ln_g, ln_b,
                                         WqS, bq, WkeS, bke, WvS, bv, WgS, bg,
                                         qWS, keWS, veWS, gateWS);

    // 3) fused triangle attention: one WG per (b,i,x)
    attn_kernel<<<NB * NN * NN, 128, 0, stream>>>(tri_emb, pair_mask,
                                                  WkaS, bka, WvaS, bva,
                                                  qWS, keWS, veWS, gateWS,
                                                  Wo, bo, (float*)d_out);
}
